// TemporalNormalization_60447369724326
// MI455X (gfx1250) — matrix operand order repaired and checked by reading, not compiled, not run on hardware
//
#include <hip/hip_runtime.h>

typedef __bf16 bf16_t;
typedef __attribute__((ext_vector_type(16))) __bf16 v16bf;
typedef __attribute__((ext_vector_type(8)))  float  v8f;
typedef unsigned int v4u __attribute__((ext_vector_type(4)));
typedef int          v8i __attribute__((ext_vector_type(8)));
typedef int          v4i __attribute__((ext_vector_type(4)));

#define D_MODEL 1024
#define SEQ     2048
#define BATCH   4
#define NHEAD   8
#define HDIM    128
#define TSTEPS  5

// ---------------- WMMA fragment helpers (CDNA5 16x16x32 bf16) ----------------
// A layout: lane<16 -> row=lane%16, K {0..7,16..23}; lane>=16 -> K {8..15,24..31}
static __device__ __forceinline__ v16bf a_frag_ld(const bf16_t* p, int l16, int half, int stride) {
  const bf16_t* q = p + l16 * stride + half * 8;
  union { uint4 u[2]; v16bf v; } c;
  c.u[0] = *reinterpret_cast<const uint4*>(q);
  c.u[1] = *reinterpret_cast<const uint4*>(q + 16);
  return c.v;
}
// B layout: lane<16 -> col=lane%16, K {0..7,16..23}; lane>=16 -> K {8..15,24..31}
static __device__ __forceinline__ v16bf b_frag_ld(const bf16_t* p, int l16, int half, int stride) {
  const bf16_t* q = p + l16 * stride + half * 8;
  union { uint4 u[2]; v16bf v; } c;
  c.u[0] = *reinterpret_cast<const uint4*>(q);
  c.u[1] = *reinterpret_cast<const uint4*>(q + 16);
  return c.v;
}
static __device__ __forceinline__ v8f wmma_bf16(v16bf a, v16bf b, v8f c) {
  return __builtin_amdgcn_wmma_f32_16x16x32_bf16(false, a, false, b, (short)0, c, false, false);
}

// ---------------- Tensor Data Mover: 2D bf16 tile -> LDS ----------------
// Packs a D# per CDNA5 ISA and issues tensor_load_to_lds
// (6-arg clang-23 form: g0, g1, g2, g3, aux v8i, cpol).
// stride0 / dims are in elements (data_size units, bf16 => data_size code 1).
// pad codes: interval 2^(code+1) DWORDs, amount (code+1) DWORDs.
static __device__ __forceinline__ unsigned lds_off_of(const void* p) {
  return (unsigned)(unsigned long)(__attribute__((address_space(3))) const void*)p;
}
static __device__ __forceinline__ void tdm_load_2d(const void* gaddr, unsigned lds_off,
                                                   unsigned tensor_d0, unsigned tensor_d1,
                                                   unsigned tile_d0, unsigned tile_d1,
                                                   unsigned long stride0,
                                                   int pad_interval_code, int pad_amount_code) {
  unsigned long ga = (unsigned long)gaddr;
  v4u g0;
  g0[0] = 1u;                                            // count=1, user mode
  g0[1] = lds_off;                                       // lds_addr (bytes)
  g0[2] = (unsigned)(ga & 0xffffffffu);                  // global_addr[31:0]
  g0[3] = (unsigned)((ga >> 32) & 0x01ffffffu) | (2u << 30); // global_addr[56:32], type=2
  v8i g1;
  g1[0] = (int)((1u << 16)                               // data_size = 2 bytes
        | (1u << 20)                                     // pad_enable
        | ((unsigned)pad_interval_code << 22)
        | ((unsigned)pad_amount_code << 25));
  g1[1] = (int)((tensor_d0 & 0xffffu) << 16);            // tensor_dim0[15:0]
  g1[2] = (int)((tensor_d0 >> 16) | ((tensor_d1 & 0xffffu) << 16));
  g1[3] = (int)((tensor_d1 >> 16) | (tile_d0 << 16));    // tile_dim0
  g1[4] = (int)(tile_d1);                                // tile_dim1, tile_dim2=0
  g1[5] = (int)(stride0 & 0xffffffffu);                  // tensor_dim0_stride[31:0]
  g1[6] = (int)((stride0 >> 32) & 0xffffu);              // stride0[47:32], stride1=0
  g1[7] = 0;
  v4i z4 = {0, 0, 0, 0};
  v8i z8 = {0, 0, 0, 0, 0, 0, 0, 0};
  __builtin_amdgcn_tensor_load_to_lds(g0, g1, z4, z4, z8, 0);
}

// ---------------- fp32 -> bf16 conversion ----------------
__global__ void cvt_kernel(const float* __restrict__ src, bf16_t* __restrict__ dst, int n) {
  int i = blockIdx.x * blockDim.x + threadIdx.x;
  if (i < n) dst[i] = (bf16_t)src[i];
}

// ---------------- indexed LayerNorm + decay ----------------
__global__ __launch_bounds__(256)
void ln_kernel(const float* __restrict__ x, const int* __restrict__ ts,
               const float* __restrict__ gamma, const float* __restrict__ beta,
               const float* __restrict__ decay, bf16_t* __restrict__ x1b) {
  __shared__ float red[256];
  __shared__ float s_mu, s_rs;
  const int row = blockIdx.x;
  const int tid = threadIdx.x;
  const float* xr = x + (size_t)row * D_MODEL;
  float v[4];
  float s = 0.f;
#pragma unroll
  for (int i = 0; i < 4; ++i) { v[i] = xr[tid + i * 256]; s += v[i]; }
  red[tid] = s; __syncthreads();
  for (int o = 128; o > 0; o >>= 1) { if (tid < o) red[tid] += red[tid + o]; __syncthreads(); }
  if (tid == 0) s_mu = red[0] * (1.0f / D_MODEL);
  __syncthreads();
  const float mu = s_mu;
  float s2 = 0.f;
#pragma unroll
  for (int i = 0; i < 4; ++i) { float d = v[i] - mu; s2 += d * d; }
  __syncthreads();
  red[tid] = s2; __syncthreads();
  for (int o = 128; o > 0; o >>= 1) { if (tid < o) red[tid] += red[tid + o]; __syncthreads(); }
  if (tid == 0) s_rs = rsqrtf(red[0] * (1.0f / D_MODEL) + 1e-5f);
  __syncthreads();
  const float rs = s_rs;
  const int t = ts[row];
  const bool use = (t < TSTEPS);
  const int ti = use ? t : (TSTEPS - 1);
  const float dk = decay[ti];
#pragma unroll
  for (int i = 0; i < 4; ++i) {
    int d = tid + i * 256;
    float g = gamma[ti * D_MODEL + d], be = beta[ti * D_MODEL + d];
    float val = use ? (((v[i] - mu) * rs) * g + be) * dk : v[i];
    x1b[(size_t)row * D_MODEL + d] = (bf16_t)val;
  }
}

// ---------------- generic WMMA GEMM: C[M,N] = A[M,K] @ W[N,K]^T (+bias) ----------------
// TDM ping/pong tile staging (compile-time LDS offsets keep the D# fully scalar);
// 8 waves each compute a 32x32 tile of the 128x64 block tile.
// mode 0: shift (+resid, fp32+bf16 out); 1: qkv split; 2: out += acc.
__global__ __launch_bounds__(256)
void gemm_bf16_kernel(const bf16_t* __restrict__ A, const bf16_t* __restrict__ W,
                      const float* __restrict__ bias, int M, int N, int K,
                      int mode, const bf16_t* __restrict__ resid,
                      float* __restrict__ outF, bf16_t* __restrict__ outB,
                      bf16_t* __restrict__ qb, bf16_t* __restrict__ kb,
                      bf16_t* __restrict__ vb, float qscale) {
  __shared__ __align__(16) bf16_t As[2][128 * 40];
  __shared__ __align__(16) bf16_t Bs[2][64 * 40];
  const int tid = threadIdx.x;
  const int lane = tid & 31, l16 = lane & 15, half = lane >> 4, w = tid >> 5;
  const int m0 = blockIdx.y * 128, n0 = blockIdx.x * 64;
  const int waveM = w >> 1, waveN = w & 1;

  const bf16_t* Ag = A + (size_t)m0 * K;
  const bf16_t* Wg = W + (size_t)n0 * K;

  // prologue: TDM prefetch of first tile pair (wave0 only; EXEC-independent, wave-level op)
  if (w == 0) {
    tdm_load_2d(Ag, lds_off_of(&As[0][0]), K, M - m0, 32, 128, (unsigned long)K, 3, 3);
    tdm_load_2d(Wg, lds_off_of(&Bs[0][0]), K, N - n0, 32, 64, (unsigned long)K, 3, 3);
  }

  v8f cf[2][2] = {};
  for (int k0 = 0; k0 < K; k0 += 64) {
    // ---- ping: compute from buffer 0, prefetch into buffer 1 ----
    __builtin_amdgcn_s_wait_tensorcnt(0);  // no-op for non-issuing waves (TENSORcnt==0)
    __syncthreads();
    if (w == 0) {                           // k0+32 < K always (K multiple of 64)
      tdm_load_2d(Ag + k0 + 32, lds_off_of(&As[1][0]), K, M - m0, 32, 128,
                  (unsigned long)K, 3, 3);
      tdm_load_2d(Wg + k0 + 32, lds_off_of(&Bs[1][0]), K, N - n0, 32, 64,
                  (unsigned long)K, 3, 3);
    }
    {
      v16bf a0 = a_frag_ld(&As[0][(waveM * 32) * 40], l16, half, 40);
      v16bf a1 = a_frag_ld(&As[0][(waveM * 32 + 16) * 40], l16, half, 40);
      v16bf b0 = b_frag_ld(&Bs[0][(waveN * 32) * 40], l16, half, 40);
      v16bf b1 = b_frag_ld(&Bs[0][(waveN * 32 + 16) * 40], l16, half, 40);
      cf[0][0] = wmma_bf16(a0, b0, cf[0][0]);
      cf[0][1] = wmma_bf16(a0, b1, cf[0][1]);
      cf[1][0] = wmma_bf16(a1, b0, cf[1][0]);
      cf[1][1] = wmma_bf16(a1, b1, cf[1][1]);
    }
    // ---- pong: compute from buffer 1, prefetch into buffer 0 ----
    __builtin_amdgcn_s_wait_tensorcnt(0);
    __syncthreads();
    if (w == 0 && (k0 + 64) < K) {
      tdm_load_2d(Ag + k0 + 64, lds_off_of(&As[0][0]), K, M - m0, 32, 128,
                  (unsigned long)K, 3, 3);
      tdm_load_2d(Wg + k0 + 64, lds_off_of(&Bs[0][0]), K, N - n0, 32, 64,
                  (unsigned long)K, 3, 3);
    }
    {
      v16bf a0 = a_frag_ld(&As[1][(waveM * 32) * 40], l16, half, 40);
      v16bf a1 = a_frag_ld(&As[1][(waveM * 32 + 16) * 40], l16, half, 40);
      v16bf b0 = b_frag_ld(&Bs[1][(waveN * 32) * 40], l16, half, 40);
      v16bf b1 = b_frag_ld(&Bs[1][(waveN * 32 + 16) * 40], l16, half, 40);
      cf[0][0] = wmma_bf16(a0, b0, cf[0][0]);
      cf[0][1] = wmma_bf16(a0, b1, cf[0][1]);
      cf[1][0] = wmma_bf16(a1, b0, cf[1][0]);
      cf[1][1] = wmma_bf16(a1, b1, cf[1][1]);
    }
  }

#pragma unroll
  for (int i = 0; i < 2; ++i)
#pragma unroll
    for (int j = 0; j < 2; ++j)
#pragma unroll
      for (int r = 0; r < 8; ++r) {
        int gm = m0 + waveM * 32 + i * 16 + half * 8 + r;
        int gn = n0 + waveN * 32 + j * 16 + l16;
        float v = cf[i][j][r] + bias[gn];
        size_t idx = (size_t)gm * N + gn;
        if (mode == 0) {
          v += (float)resid[idx];
          outF[idx] = v;
          outB[idx] = (bf16_t)v;
        } else if (mode == 1) {
          if (gn < D_MODEL)           qb[(size_t)gm * D_MODEL + gn] = (bf16_t)(v * qscale);
          else if (gn < 2 * D_MODEL)  kb[(size_t)gm * D_MODEL + (gn - D_MODEL)] = (bf16_t)v;
          else                        vb[(size_t)gm * D_MODEL + (gn - 2 * D_MODEL)] = (bf16_t)v;
        } else {
          outF[idx] += v;
        }
      }
}

// ---------------- flash attention: 128 q-rows / block, 32-key tiles ----------------
// K tile staged by TDM (overlaps the VGPR-path V transpose scatter, which TDM can't do).
__global__ __launch_bounds__(256)
void attn_kernel(const bf16_t* __restrict__ qg, const bf16_t* __restrict__ kg,
                 const bf16_t* __restrict__ vg, bf16_t* __restrict__ ctx) {
  __shared__ __align__(16) bf16_t Ks[32 * 136];    // K tile [token][feat], stride 136
  __shared__ __align__(16) bf16_t VT[128 * 40];    // V tile transposed [feat][token]
  __shared__ __align__(16) bf16_t Ps[8 * 16 * 40]; // per-wave P tile [row][key]
  const int tid = threadIdx.x;
  const int lane = tid & 31, l16 = lane & 15, half = lane >> 4, w = tid >> 5;
  const int q0 = blockIdx.x * 128;
  const int h = blockIdx.y, b = blockIdx.z;
  const size_t rowbase = (size_t)b * SEQ;
  const int colbase = h * HDIM;
  const int qrow = q0 + w * 16;

  v16bf qf[4];
#pragma unroll
  for (int c = 0; c < 4; ++c)
    qf[c] = a_frag_ld(qg + (rowbase + qrow) * D_MODEL + colbase + 32 * c, l16, half, D_MODEL);

  v8f acc[8] = {};
  float mrow[8], lrow[8];
#pragma unroll
  for (int r = 0; r < 8; ++r) { mrow[r] = -1e30f; lrow[r] = 0.f; }
  bf16_t* Pw = Ps + w * 16 * 40;
  const unsigned ksOff = lds_off_of(&Ks[0]);

  for (int n0 = 0; n0 < SEQ; n0 += 32) {
    __syncthreads();  // previous compute done: LDS tiles reusable
    if (w == 0) {     // K tile 32x128 via TDM (row 128 bf16 = 64 DW -> interval code 5; pad 4 DW -> code 3)
      tdm_load_2d(kg + (rowbase + n0) * D_MODEL + colbase, ksOff,
                  D_MODEL, (unsigned)(SEQ - n0), 128, 32, (unsigned long)D_MODEL, 5, 3);
    }
#pragma unroll
    for (int i = 0; i < 2; ++i) {  // V tile transposed (VGPR path)
      int c = tid + i * 256;
      int tok = c >> 4, kc = c & 15;
      union { uint4 u; bf16_t h8[8]; } ld;
      ld.u = *reinterpret_cast<const uint4*>(&vg[(rowbase + n0 + tok) * D_MODEL + colbase + kc * 8]);
#pragma unroll
      for (int jj = 0; jj < 8; ++jj)
        VT[(kc * 8 + jj) * 40 + tok] = ld.h8[jj];
    }
    __builtin_amdgcn_s_wait_tensorcnt(0);  // no-op for waves that issued nothing
    __syncthreads();

    // S = Q K^T (softmax scale folded into Q)
    v8f c0 = {}, c1 = {};
#pragma unroll
    for (int c = 0; c < 4; ++c) {
      v16bf bk0 = b_frag_ld(Ks + 32 * c, l16, half, 136);
      v16bf bk1 = b_frag_ld(Ks + 16 * 136 + 32 * c, l16, half, 136);
      c0 = wmma_bf16(qf[c], bk0, c0);
      c1 = wmma_bf16(qf[c], bk1, c1);
    }
    // online softmax per row (rows r / r+8 across lane halves)
#pragma unroll
    for (int r = 0; r < 8; ++r) {
      float s0 = c0[r], s1 = c1[r];
      float t = fmaxf(s0, s1);
#pragma unroll
      for (int off = 8; off > 0; off >>= 1) t = fmaxf(t, __shfl_xor(t, off, 32));
      float mnew = fmaxf(mrow[r], t);
      float alpha = __expf(mrow[r] - mnew);
      mrow[r] = mnew;
      float p0 = __expf(s0 - mnew);
      float p1 = __expf(s1 - mnew);
      float ps = p0 + p1;
#pragma unroll
      for (int off = 8; off > 0; off >>= 1) ps += __shfl_xor(ps, off, 32);
      lrow[r] = lrow[r] * alpha + ps;
#pragma unroll
      for (int f = 0; f < 8; ++f) acc[f][r] *= alpha;
      Pw[(half * 8 + r) * 40 + l16]      = (bf16_t)p0;
      Pw[(half * 8 + r) * 40 + 16 + l16] = (bf16_t)p1;
    }
    // O += P V  (P re-fragmented via LDS; same-wave DS ops are in-order)
    v16bf pa = a_frag_ld(Pw, l16, half, 40);
#pragma unroll
    for (int f = 0; f < 8; ++f) {
      v16bf bv = b_frag_ld(VT + 16 * f * 40, l16, half, 40);
      acc[f] = wmma_bf16(pa, bv, acc[f]);
    }
  }

#pragma unroll
  for (int r = 0; r < 8; ++r) {
    float rinv = 1.0f / lrow[r];
    int grow = qrow + half * 8 + r;
#pragma unroll
    for (int f = 0; f < 8; ++f)
      ctx[(rowbase + grow) * D_MODEL + colbase + 16 * f + l16] = (bf16_t)(acc[f][r] * rinv);
  }
}

// ---------------- launch ----------------
extern "C" void kernel_launch(void* const* d_in, const int* in_sizes, int n_in,
                              void* d_out, int out_size, void* d_ws, size_t ws_size,
                              hipStream_t stream) {
  (void)in_sizes; (void)n_in; (void)out_size; (void)ws_size;
  const float* x      = (const float*)d_in[0];
  const int*   ts     = (const int*)d_in[1];
  const float* gamma  = (const float*)d_in[2];
  const float* beta   = (const float*)d_in[3];
  const float* decay  = (const float*)d_in[4];
  const float* shiftW = (const float*)d_in[5];
  const float* shiftB = (const float*)d_in[6];
  const float* inW    = (const float*)d_in[7];
  const float* inB    = (const float*)d_in[8];
  const float* outW   = (const float*)d_in[9];
  const float* outB   = (const float*)d_in[10];
  float* out = (float*)d_out;

  const size_t MR = (size_t)BATCH * SEQ; // 8192 token rows
  char* p = (char*)d_ws;
  bf16_t* wShift = (bf16_t*)p; p += (size_t)D_MODEL * D_MODEL * 2;
  bf16_t* wIn    = (bf16_t*)p; p += (size_t)3 * D_MODEL * D_MODEL * 2;
  bf16_t* wOut   = (bf16_t*)p; p += (size_t)D_MODEL * D_MODEL * 2;
  bf16_t* x1b    = (bf16_t*)p; p += MR * D_MODEL * 2;  // reused as ctx later
  bf16_t* x2b    = (bf16_t*)p; p += MR * D_MODEL * 2;
  bf16_t* qb     = (bf16_t*)p; p += MR * D_MODEL * 2;
  bf16_t* kb     = (bf16_t*)p; p += MR * D_MODEL * 2;
  bf16_t* vb     = (bf16_t*)p; p += MR * D_MODEL * 2;
  bf16_t* ctxb   = x1b;

  const int nw = D_MODEL * D_MODEL;
  cvt_kernel<<<(nw + 255) / 256, 256, 0, stream>>>(shiftW, wShift, nw);
  cvt_kernel<<<(3 * nw + 255) / 256, 256, 0, stream>>>(inW, wIn, 3 * nw);
  cvt_kernel<<<(nw + 255) / 256, 256, 0, stream>>>(outW, wOut, nw);

  ln_kernel<<<(int)MR, 256, 0, stream>>>(x, ts, gamma, beta, decay, x1b);

  const float qscale = 0.088388347648318447f; // 1/sqrt(128)

  // x2 = x1 + x1 @ shiftW^T + b  -> d_out (fp32) + bf16 copy
  gemm_bf16_kernel<<<dim3(D_MODEL / 64, (int)(MR / 128)), 256, 0, stream>>>(
      x1b, wShift, shiftB, (int)MR, D_MODEL, D_MODEL, 0, x1b, out, x2b,
      nullptr, nullptr, nullptr, 1.0f);

  // qkv = x2 @ inW^T + b -> q (scaled), k, v
  gemm_bf16_kernel<<<dim3(3 * D_MODEL / 64, (int)(MR / 128)), 256, 0, stream>>>(
      x2b, wIn, inB, (int)MR, 3 * D_MODEL, D_MODEL, 1, nullptr, nullptr, nullptr,
      qb, kb, vb, qscale);

  // attention -> ctx (bf16)
  attn_kernel<<<dim3(SEQ / 128, NHEAD, BATCH), 256, 0, stream>>>(qb, kb, vb, ctxb);

  // d_out += ctx @ outW^T + b
  gemm_bf16_kernel<<<dim3(D_MODEL / 64, (int)(MR / 128)), 256, 0, stream>>>(
      ctxb, wOut, outB, (int)MR, D_MODEL, D_MODEL, 2, nullptr, out, nullptr,
      nullptr, nullptr, nullptr, 1.0f);
}
